// MultiHeadAttention_9912784519532
// MI455X (gfx1250) — matrix operand order repaired and checked by reading, not compiled, not run on hardware
//
#include <hip/hip_runtime.h>

typedef __attribute__((ext_vector_type(16))) _Float16 v16h;
typedef __attribute__((ext_vector_type(8)))  _Float16 v8h;
typedef __attribute__((ext_vector_type(8)))  float    v8f;

#define HEADS  8
#define DHEAD  64
#define SEQ    2048
#define BATCH  2
#define DIM    512
#define MAXPOS 512
#define NREL   1025
#define NRELP  1040   // padded rel rows
#define LDP    1040   // padded P row stride (65 tiles * 16)

// ---------- fragment loaders (layouts per CDNA5 ISA 7.12.2) ----------

// A-fragment 16x32 f16 from row-major tile. Lane l<16: row l, K = 0..7 & 16..23;
// lane l+16: row l, K = 8..15 & 24..31. VGPR halves K-ascending.
__device__ __forceinline__ v16h load_a_f16(const _Float16* base, int ld) {
  int lane = threadIdx.x & 31;
  int r = lane & 15, cb = (lane >> 4) << 3;
  const _Float16* p = base + (long)r * ld + cb;
  union { v16h v; struct { v8h lo; v8h hi; } s; } u;
  u.s.lo = *(const v8h*)(p);
  u.s.hi = *(const v8h*)(p + 16);
  return u.v;
}

// B-fragment 32x16 given the row-major B^T tile (16 x 32): lanes 0-15 hold
// K=0..7 & 16..23, lanes 16-31 hold K=8..15 & 24..31, N = lane&15.
__device__ __forceinline__ v16h load_bt_f16(const _Float16* baseT, int ld) {
  int lane = threadIdx.x & 31;
  int nn = lane & 15, kb = (lane >> 4) << 3;
  const _Float16* p = baseT + (long)nn * ld + kb;
  union { v16h v; struct { v8h lo; v8h hi; } s; } u;
  u.s.lo = *(const v8h*)(p);
  u.s.hi = *(const v8h*)(p + 16);
  return u.v;
}

__device__ __forceinline__ v8f wmma_f16(v16h a, v16h b, v8f c) {
  return __builtin_amdgcn_wmma_f32_16x16x32_f16(false, a, false, b, (short)0, c,
                                                false, false);
}

// 16-byte async global -> LDS copy (per lane), tracked by ASYNCcnt
__device__ __forceinline__ void async_copy16(const void* g, unsigned lds) {
  asm volatile("global_load_async_to_lds_b128 %0, %1, off"
               :: "v"(lds), "v"(g) : "memory");
}

// ---------- prep kernels: one-time f32 -> f16 conversions / transposes ----------

__global__ __launch_bounds__(256) void cvt_f32_f16_kernel(
    const float* __restrict__ src, _Float16* __restrict__ dst, int n) {
  int i = blockIdx.x * 256 + threadIdx.x;
  if (i < n) dst[i] = (_Float16)src[i];
}

// WT[n][k] = W[k][n], f32 -> f16
__global__ __launch_bounds__(256) void transpose_cvt_kernel(
    const float* __restrict__ W, _Float16* __restrict__ WT, int K, int N) {
  int i = blockIdx.x * 256 + threadIdx.x;
  if (i < K * N) {
    int n = i / K, k = i % K;
    WT[i] = (_Float16)W[(long)k * N + n];
  }
}

// relh: [1040][64] f16, rows >= 1025 zeroed
__global__ __launch_bounds__(256) void cvt_rel_kernel(
    const float* __restrict__ rel, _Float16* __restrict__ relh) {
  int i = blockIdx.x * 256 + threadIdx.x;
  if (i < NRELP * DHEAD) {
    int r = i >> 6, c = i & 63;
    relh[i] = (r < NREL) ? (_Float16)rel[(long)r * DHEAD + c] : (_Float16)0.f;
  }
}

// ---------- kernel 1: QKV projection, 16x64 strip per wave, SW-pipelined ----------

__global__ __launch_bounds__(256) void qkv_proj_kernel(
    const _Float16* __restrict__ xh, const _Float16* __restrict__ WqT,
    const _Float16* __restrict__ WkvT,
    _Float16* __restrict__ Qs, _Float16* __restrict__ Kh, _Float16* __restrict__ Vt)
{
  const int ROWT = (BATCH * SEQ) / 16;   // 256
  const int CGRP = 8 + 16;               // 64-col groups: 8 for Q, 16 for KV
  int job = blockIdx.x * 8 + (threadIdx.x >> 5);
  if (job >= ROWT * CGRP) return;        // uniform per wave
  int mt = job % ROWT;
  int cg = job / ROWT;

  const _Float16* WT; int col0;
  if (cg < 8) { WT = WqT;  col0 = cg * 64; }
  else        { WT = WkvT; col0 = (cg - 8) * 64; }

  const _Float16* abase = xh + (long)mt * 16 * DIM;

  v16h a = load_a_f16(abase, DIM);
  v16h b[4];
#pragma unroll
  for (int t = 0; t < 4; ++t)
    b[t] = load_bt_f16(WT + (long)(col0 + t * 16) * DIM, DIM);

  v8f acc[4] = {};
#pragma unroll 4
  for (int kk = 0; kk < DIM; kk += 32) {
    int kn = (kk + 32 < DIM) ? kk + 32 : kk;          // prefetch next (or reload last)
    v16h an = load_a_f16(abase + kn, DIM);
    v16h bn[4];
#pragma unroll
    for (int t = 0; t < 4; ++t)
      bn[t] = load_bt_f16(WT + (long)(col0 + t * 16) * DIM + kn, DIM);
#pragma unroll
    for (int t = 0; t < 4; ++t)
      acc[t] = wmma_f16(a, b[t], acc[t]);
    a = an;
#pragma unroll
    for (int t = 0; t < 4; ++t) b[t] = bn[t];
  }

  int lane = threadIdx.x & 31;
  int col = lane & 15, rb = (lane >> 4) << 3;
#pragma unroll
  for (int t = 0; t < 4; ++t) {
#pragma unroll
    for (int r = 0; r < 8; ++r) {
      int grow = mt * 16 + rb + r;
      int bb = grow >> 11, i = grow & (SEQ - 1);
      float v = acc[t][r];
      int dd = t * 16 + col;   // 64-col groups align with one head
      if (cg < 8) {
        int hh = cg;
        Qs[(((long)(bb * HEADS + hh)) * SEQ + i) * DHEAD + dd] = (_Float16)(v * 0.125f);
      } else if (cg < 16) {
        int hh = cg - 8;
        Kh[(((long)(bb * HEADS + hh)) * SEQ + i) * DHEAD + dd] = (_Float16)v;
      } else {
        int hh = cg - 16;
        Vt[(((long)(bb * HEADS + hh)) * DHEAD + dd) * SEQ + i] = (_Float16)v;  // transposed
      }
    }
  }
}

// ---------- kernel 2: P = Q @ rel_emb^T  (skew trick for relative pos) ----------

__global__ __launch_bounds__(256) void relpos_kernel(
    const _Float16* __restrict__ Qs, const _Float16* __restrict__ relh,
    _Float16* __restrict__ P)
{
  const int ROWT = (BATCH * HEADS * SEQ) / 16;  // 2048
  const int COLT = NRELP / 16;                  // 65
  int job = blockIdx.x * 8 + (threadIdx.x >> 5);
  if (job >= ROWT * COLT) return;
  int mt = job % ROWT;
  int ct = job / ROWT;
  int c0 = ct * 16;

  v8f acc = {};
#pragma unroll
  for (int kk = 0; kk < DHEAD; kk += 32) {
    v16h a = load_a_f16(Qs + (long)mt * 16 * DHEAD + kk, DHEAD);
    v16h b = load_bt_f16(relh + (long)c0 * DHEAD + kk, DHEAD);  // padded rows are 0
    acc = wmma_f16(a, b, acc);
  }

  int lane = threadIdx.x & 31;
  int col = lane & 15, rb = (lane >> 4) << 3;
#pragma unroll
  for (int r = 0; r < 8; ++r)
    P[((long)(mt * 16 + rb + r)) * LDP + c0 + col] = (_Float16)acc[r];
}

// ---------- kernel 3: flash attention, async K/V staging through LDS ----------

__global__ __launch_bounds__(128) void attn_kernel(
    const _Float16* __restrict__ Qs, const _Float16* __restrict__ Kh,
    const _Float16* __restrict__ Vt, const _Float16* __restrict__ P,
    _Float16* __restrict__ AO)
{
  // double-buffered cooperative K/V tiles + per-wave prob staging
  __shared__ _Float16 lkbuf[2][32 * DHEAD];   // [keys 32][d 64]   4KB each
  __shared__ _Float16 lvbuf[2][DHEAD * 32];   // [d 64][keys 32]   4KB each
  __shared__ _Float16 pbuf[4][16 * 32];       // per-wave prob tile

  int tid  = threadIdx.x;                 // 0..127
  int wave = tid >> 5;
  int qt = blockIdx.x * 4 + wave;         // 2048 q-tiles; all 4 waves share bh
  int bh = qt >> 7;                       // (b*8+h)
  int mt = qt & 127;
  int i0 = mt * 16;
  int lane = tid & 31;
  int col = lane & 15, rb = (lane >> 4) << 3;

  const _Float16* qbase = Qs + ((long)bh * SEQ + i0) * DHEAD;
  v16h qa0 = load_a_f16(qbase, DHEAD);        // K(dim) 0..31
  v16h qa1 = load_a_f16(qbase + 32, DHEAD);   // K(dim) 32..63

  const _Float16* kbase = Kh + (long)bh * SEQ * DHEAD;
  const _Float16* vbase = Vt + (long)bh * DHEAD * SEQ;
  const _Float16* prow  = P + (long)(bh * SEQ) * LDP;

  unsigned lk_lds[2], lv_lds[2];
  lk_lds[0] = (unsigned)(unsigned long long)(void*)&lkbuf[0][0];
  lk_lds[1] = (unsigned)(unsigned long long)(void*)&lkbuf[1][0];
  lv_lds[0] = (unsigned)(unsigned long long)(void*)&lvbuf[0][0];
  lv_lds[1] = (unsigned)(unsigned long long)(void*)&lvbuf[1][0];

  // stage K(32x64) + Vt(64x32) tiles for chunk starting at j0 into buffer `buf`
  auto stage_kv = [&](int j0, int buf) {
    const char* kg = (const char*)(kbase + (long)j0 * DHEAD);  // 4KB contiguous
    async_copy16(kg + tid * 16,          lk_lds[buf] + tid * 16);
    async_copy16(kg + (tid + 128) * 16,  lk_lds[buf] + (tid + 128) * 16);
    const char* vg = (const char*)(vbase + j0);                // 64 rows x 64B, stride SEQ*2
    int c0 = tid, c1 = tid + 128;
    async_copy16(vg + (long)(c0 >> 2) * (SEQ * 2) + (c0 & 3) * 16, lv_lds[buf] + c0 * 16);
    async_copy16(vg + (long)(c1 >> 2) * (SEQ * 2) + (c1 & 3) * 16, lv_lds[buf] + c1 * 16);
  };

  v8f o0 = {}, o1 = {}, o2 = {}, o3 = {};
  float mrow[8], lrow[8];
#pragma unroll
  for (int r = 0; r < 8; ++r) { mrow[r] = -3.0e38f; lrow[r] = 0.f; }

  _Float16* pb = pbuf[wave];

  stage_kv(0, 0);                         // prologue: fill buffer 0

  for (int j0 = 0; j0 < SEQ; j0 += 32) {
    int cur = (j0 >> 5) & 1;
    if (j0 + 32 < SEQ) {
      stage_kv(j0 + 32, cur ^ 1);         // issue next stage (4 async ops/wave)
      asm volatile("s_wait_asynccnt 0x4" ::: "memory");  // current stage done
    } else {
      asm volatile("s_wait_asynccnt 0x0" ::: "memory");
    }
    __syncthreads();                      // all waves' stages visible

    const _Float16* lk = lkbuf[cur];      // [32][64]
    const _Float16* lv = lvbuf[cur];      // [64][32]

    // ---- scores: Q(16x64) @ K^T, two 16-key tiles (B frags from LDS) ----
    v8f s0 = {}, s1 = {};
    s0 = wmma_f16(qa0, load_bt_f16(lk, DHEAD), s0);
    s0 = wmma_f16(qa1, load_bt_f16(lk + 32, DHEAD), s0);
    s1 = wmma_f16(qa0, load_bt_f16(lk + 16 * DHEAD, DHEAD), s1);
    s1 = wmma_f16(qa1, load_bt_f16(lk + 16 * DHEAD + 32, DHEAD), s1);

    // ---- add relative-position bias: pos[i,j] = P[i, clip(i-j)+512] ----
#pragma unroll
    for (int r = 0; r < 8; ++r) {
      int i = i0 + rb + r;
      int j = j0 + col;
      int d0 = i - j;        d0 = d0 < -MAXPOS ? -MAXPOS : (d0 > MAXPOS ? MAXPOS : d0);
      int d1 = i - (j + 16); d1 = d1 < -MAXPOS ? -MAXPOS : (d1 > MAXPOS ? MAXPOS : d1);
      s0[r] += (float)prow[(long)i * LDP + d0 + MAXPOS];
      s1[r] += (float)prow[(long)i * LDP + d1 + MAXPOS];
    }

    // ---- online softmax over this 32-key chunk ----
    float p0[8], p1[8];
#pragma unroll
    for (int r = 0; r < 8; ++r) {
      float cm = fmaxf(s0[r], s1[r]);
      cm = fmaxf(cm, __shfl_xor(cm, 1, 32));
      cm = fmaxf(cm, __shfl_xor(cm, 2, 32));
      cm = fmaxf(cm, __shfl_xor(cm, 4, 32));
      cm = fmaxf(cm, __shfl_xor(cm, 8, 32));
      float mnew = fmaxf(mrow[r], cm);
      float fsc = __expf(mrow[r] - mnew);
      mrow[r] = mnew;
      p0[r] = __expf(s0[r] - mnew);
      p1[r] = __expf(s1[r] - mnew);
      float rs = p0[r] + p1[r];
      rs += __shfl_xor(rs, 1, 32);
      rs += __shfl_xor(rs, 2, 32);
      rs += __shfl_xor(rs, 4, 32);
      rs += __shfl_xor(rs, 8, 32);
      lrow[r] = lrow[r] * fsc + rs;
      o0[r] *= fsc; o1[r] *= fsc; o2[r] *= fsc; o3[r] *= fsc;
      int row = rb + r;
      pb[row * 32 + col]      = (_Float16)p0[r];
      pb[row * 32 + 16 + col] = (_Float16)p1[r];
    }
    asm volatile("s_wait_dscnt 0x0" ::: "memory");
    v16h pa = load_a_f16(pb, 32);           // probs as A fragment

    // ---- O += probs(16x32) @ V(32x64), B frags from LDS Vt tile ----
    o0 = wmma_f16(pa, load_bt_f16(lv + 0 * 16 * 32, 32), o0);
    o1 = wmma_f16(pa, load_bt_f16(lv + 1 * 16 * 32, 32), o1);
    o2 = wmma_f16(pa, load_bt_f16(lv + 2 * 16 * 32, 32), o2);
    o3 = wmma_f16(pa, load_bt_f16(lv + 3 * 16 * 32, 32), o3);

    __syncthreads();                      // all reads done before buffer reuse
  }

  // ---- epilogue: normalize, store [b][n][h*d] f16 for the output GEMM ----
  int bb = bh >> 3, hh = bh & 7;
#pragma unroll
  for (int r = 0; r < 8; ++r) {
    float inv = 1.0f / lrow[r];
    long row = (long)bb * SEQ + i0 + rb + r;
    _Float16* op = AO + row * (HEADS * DHEAD) + hh * DHEAD;
    op[0 * 16 + col] = (_Float16)(o0[r] * inv);
    op[1 * 16 + col] = (_Float16)(o1[r] * inv);
    op[2 * 16 + col] = (_Float16)(o2[r] * inv);
    op[3 * 16 + col] = (_Float16)(o3[r] * inv);
  }
}

// ---------- kernel 4: output projection, 16x64 strip per wave, SW-pipelined ----------

__global__ __launch_bounds__(256) void oproj_kernel(
    const _Float16* __restrict__ AO, const _Float16* __restrict__ WoT,
    const float* __restrict__ bo, float* __restrict__ out)
{
  const int ROWT = (BATCH * SEQ) / 16;  // 256
  const int CGRP = DIM / 64;            // 8
  int job = blockIdx.x * 8 + (threadIdx.x >> 5);
  if (job >= ROWT * CGRP) return;
  int mt = job % ROWT, cg = job / ROWT;
  int n0 = cg * 64;

  const _Float16* abase = AO + (long)mt * 16 * DIM;

  v16h a = load_a_f16(abase, DIM);
  v16h b[4];
#pragma unroll
  for (int t = 0; t < 4; ++t)
    b[t] = load_bt_f16(WoT + (long)(n0 + t * 16) * DIM, DIM);

  v8f acc[4] = {};
#pragma unroll 4
  for (int kk = 0; kk < DIM; kk += 32) {
    int kn = (kk + 32 < DIM) ? kk + 32 : kk;
    v16h an = load_a_f16(abase + kn, DIM);
    v16h bn[4];
#pragma unroll
    for (int t = 0; t < 4; ++t)
      bn[t] = load_bt_f16(WoT + (long)(n0 + t * 16) * DIM + kn, DIM);
#pragma unroll
    for (int t = 0; t < 4; ++t)
      acc[t] = wmma_f16(a, b[t], acc[t]);
    a = an;
#pragma unroll
    for (int t = 0; t < 4; ++t) b[t] = bn[t];
  }

  int lane = threadIdx.x & 31;
  int col = lane & 15, rb = (lane >> 4) << 3;
#pragma unroll
  for (int t = 0; t < 4; ++t) {
    float bias = bo[n0 + t * 16 + col];
#pragma unroll
    for (int r = 0; r < 8; ++r)
      out[((long)mt * 16 + rb + r) * DIM + n0 + t * 16 + col] = acc[t][r] + bias;
  }
}

// ---------- launch ----------

extern "C" void kernel_launch(void* const* d_in, const int* in_sizes, int n_in,
                              void* d_out, int out_size, void* d_ws, size_t ws_size,
                              hipStream_t stream) {
  const float* x   = (const float*)d_in[0];
  const float* Wq  = (const float*)d_in[1];
  const float* Wkv = (const float*)d_in[2];
  const float* Wo  = (const float*)d_in[3];
  const float* bo  = (const float*)d_in[4];
  const float* rel = (const float*)d_in[5];
  float* out = (float*)d_out;

  char* ws = (char*)d_ws;
  _Float16* xh   = (_Float16*)(ws + (0ull  << 20));  // 4 MiB [b*n, 512]
  _Float16* Qs   = (_Float16*)(ws + (4ull  << 20));  // 4 MiB [b,h,n,d] (scaled)
  _Float16* Kh   = (_Float16*)(ws + (8ull  << 20));  // 4 MiB [b,h,n,d]
  _Float16* Vt   = (_Float16*)(ws + (12ull << 20));  // 4 MiB [b,h,d,n]
  _Float16* AO   = (_Float16*)(ws + (16ull << 20));  // 4 MiB [b*n, h*d]
  _Float16* WqT  = (_Float16*)(ws + (20ull << 20));  // 0.5 MiB [512][512]
  _Float16* WkvT = (_Float16*)(ws + (21ull << 20));  // 1 MiB [1024][512]
  _Float16* WoT  = (_Float16*)(ws + (22ull << 20));  // 0.5 MiB [512][512]
  _Float16* relh = (_Float16*)(ws + (23ull << 20));  // 130 KiB [1040][64]
  _Float16* P    = (_Float16*)(ws + (24ull << 20));  // 65 MiB [b*h*n, 1040]

  // prep: f16 conversions + weight transposes (bandwidth-trivial)
  cvt_f32_f16_kernel<<<(BATCH*SEQ*DIM + 255)/256, 256, 0, stream>>>(x, xh, BATCH*SEQ*DIM);
  transpose_cvt_kernel<<<(DIM*DIM   + 255)/256, 256, 0, stream>>>(Wq,  WqT,  DIM, DIM);
  transpose_cvt_kernel<<<(DIM*2*DIM + 255)/256, 256, 0, stream>>>(Wkv, WkvT, DIM, 2*DIM);
  transpose_cvt_kernel<<<(DIM*DIM   + 255)/256, 256, 0, stream>>>(Wo,  WoT,  DIM, DIM);
  cvt_rel_kernel<<<(NRELP*DHEAD + 255)/256, 256, 0, stream>>>(rel, relh);

  // main pipeline
  qkv_proj_kernel<<<768, 256, 0, stream>>>(xh, WqT, WkvT, Qs, Kh, Vt);
  relpos_kernel<<<16640, 256, 0, stream>>>(Qs, relh, P);
  attn_kernel<<<512, 128, 0, stream>>>(Qs, Kh, Vt, P, AO);
  oproj_kernel<<<256, 256, 0, stream>>>(AO, WoT, bo, out);
}
